// Quantizer_47450798686289
// MI455X (gfx1250) — matrix operand order repaired and checked by reading, not compiled, not run on hardware
//
#include <hip/hip_runtime.h>

// ---------------------------------------------------------------------------
// VQ-VAE quantizer for MI455X (gfx1250, wave32, WMMA).
//   z      : [32,32,32,256] f32  -> N=32768 rows, D=256
//   weight : [256,1024]      f32 -> K=1024 codewords (columns)
// out = quantized (gather of argmin codeword) transposed to [B,D,H,W],
//       + commitment_loss + embedding_loss (equal forward values).
// GEMM via split-f16 (hi/lo) v_wmma_f32_16x16x32_f16, 3 products ~ fp32.
// ---------------------------------------------------------------------------

typedef __attribute__((ext_vector_type(16))) _Float16 v16h;
typedef __attribute__((ext_vector_type(8)))  float    v8f;

#define B_   32
#define H_   32
#define W_   32
#define D_   256
#define K_   1024
#define N_   (B_*H_*W_)     // 32768 rows
#define NTILES (N_/16)      // 2048 row tiles
#define CTILES (K_/16)      // 64 code tiles
#define KSTEPS (D_/32)      // 8 k-steps of 32

#define WAVES_PER_BLOCK 8
#define MAIN_BLOCKS (NTILES/WAVES_PER_BLOCK)   // 256

// workspace layout (bytes):
//   [0, 1MB)            swizzled f16 weight hi/lo: [ct][ks][2][lane][16] halves
//   [1MB, 1MB+4KB)      ek2[1024] f32
//   [1MB+4KB, +1KB)     partials[MAIN_BLOCKS] f32
#define WS_EK2_OFF   (1u << 20)
#define WS_PART_OFF  ((1u << 20) + 4096u)

// ---------------------------------------------------------------------------
// Prep: swizzle weight into per-lane WMMA-B order (hi/lo f16 split).
// B layout (16x16x32 f16): lane n<16 holds col n, K={0..7,16..23}; lane n>=16
// holds col n-16, K={8..15,24..31}; half i: K = (i<8 ? 8h+i : 16+8h+i-8).
// One thread per (ct, ks, lane, i): 64*8*32*16 = 262144 threads.
// ---------------------------------------------------------------------------
__global__ void vq_prep_wswz(const float* __restrict__ weight,
                             _Float16* __restrict__ wswz) {
    unsigned id   = blockIdx.x * 256u + threadIdx.x;
    unsigned i    = id & 15u;
    unsigned lane = (id >> 4) & 31u;
    unsigned ks   = (id >> 9) & 7u;
    unsigned ct   = id >> 12;                 // 0..63
    unsigned n    = lane & 15u;
    unsigned d    = ks * 32u + (lane >> 4) * 8u + ((i < 8u) ? i : (8u + i));
    float w = weight[d * K_ + ct * 16u + n];
    _Float16 h = (_Float16)w;
    float lo = w - (float)h;
    unsigned base = ((ct * 8u + ks) * 2u) * 512u + lane * 16u + i;
    wswz[base]        = h;                    // hi slot
    wswz[base + 512u] = (_Float16)lo;         // lo slot
}

// ek2[k] = sum_d weight[d][k]^2     (1024 threads)
__global__ void vq_prep_ek2(const float* __restrict__ weight,
                            float* __restrict__ ek2) {
    unsigned k = blockIdx.x * 256u + threadIdx.x;
    float s = 0.f;
    for (int d = 0; d < D_; ++d) {
        float w = weight[d * K_ + k];
        s += w * w;
    }
    ek2[k] = s;
}

// ---------------------------------------------------------------------------
// Main fused kernel: per wave -> one 16-row tile of z.
//   * load z tile once (fp32), split into f16 hi/lo in WMMA-A register layout
//   * for each of 64 code tiles: 8 k-steps x 3 WMMA (hi*hi + lo*hi + hi*lo)
//   * dist = ek2[n] - 2*S ; running min/argmin per C slot
//   * cross-lane min reduce (index tie-break = first min, like jnp.argmin)
//   * loss via  sum ||q-z||^2 = sum_rows ( ||z_row||^2 + min_dist_row )
//   * gather weight columns, write [B,D,H,W] output (16-float coalesced rows)
// ---------------------------------------------------------------------------
__global__ void __launch_bounds__(256)
vq_main(const float* __restrict__ z,
        const float* __restrict__ weight,
        const _Float16* __restrict__ wswz,
        const float* __restrict__ ek2,
        float* __restrict__ out,
        float* __restrict__ partials) {
    __shared__ int   closestLds[WAVES_PER_BLOCK * 16];
    __shared__ float waveLoss[WAVES_PER_BLOCK];

    const int lane = threadIdx.x & 31;
    const int wid  = threadIdx.x >> 5;
    const int tile = blockIdx.x * WAVES_PER_BLOCK + wid;
    const int nloc = lane & 15;                    // col-in-tile / row-in-tile

    // ---- load A tile (16 rows x 256 k) as f16 hi/lo in WMMA-A layout ----
    // A layout: lane m (m<16) holds row m, K = {0..7, 16..23} per 32-k step;
    //           lane m+16 holds row m,     K = {8..15, 24..31}.
    const int row = tile * 16 + nloc;
    const float4* zr = (const float4*)(z + (size_t)row * D_);
    const int o4 = (lane < 16) ? 0 : 2;            // float4 offset (0 or 8 floats)

    v16h ah[KSTEPS], al[KSTEPS];
    float zsq = 0.f;
#pragma unroll
    for (int ks = 0; ks < KSTEPS; ++ks) {
        float4 c0 = zr[ks * 8 + o4];
        float4 c1 = zr[ks * 8 + o4 + 1];
        float4 c2 = zr[ks * 8 + 4 + o4];
        float4 c3 = zr[ks * 8 + 4 + o4 + 1];
        float vv[16] = {c0.x, c0.y, c0.z, c0.w, c1.x, c1.y, c1.z, c1.w,
                        c2.x, c2.y, c2.z, c2.w, c3.x, c3.y, c3.z, c3.w};
#pragma unroll
        for (int i = 0; i < 16; ++i) {
            float f = vv[i];
            zsq += f * f;
            _Float16 hv = (_Float16)f;
            ah[ks][i] = hv;
            al[ks][i] = (_Float16)(f - (float)hv);
        }
    }

    // ---- sweep the 64 code tiles, tracking running min per C slot ----
    float runmin[8];
    int   runidx[8];
#pragma unroll
    for (int r = 0; r < 8; ++r) { runmin[r] = 3.402823466e38f; runidx[r] = 0; }

    for (int ct = 0; ct < CTILES; ++ct) {
        float ekn = ek2[ct * 16 + nloc];
        const v16h* bp = (const v16h*)wswz + (size_t)(ct * KSTEPS) * 2 * 32;
        v8f acc = {};
#pragma unroll
        for (int ks = 0; ks < KSTEPS; ++ks) {
            v16h bh = bp[(ks * 2 + 0) * 32 + lane];
            v16h bl = bp[(ks * 2 + 1) * 32 + lane];
            acc = __builtin_amdgcn_wmma_f32_16x16x32_f16(
                false, ah[ks], false, bh, (short)0, acc, false, false);
            acc = __builtin_amdgcn_wmma_f32_16x16x32_f16(
                false, al[ks], false, bh, (short)0, acc, false, false);
            acc = __builtin_amdgcn_wmma_f32_16x16x32_f16(
                false, ah[ks], false, bl, (short)0, acc, false, false);
        }
        const int cidx = ct * 16 + nloc;
#pragma unroll
        for (int r = 0; r < 8; ++r) {
            float dist = ekn - 2.0f * acc[r];     // ||z||^2 is rank-constant
            if (dist < runmin[r]) { runmin[r] = dist; runidx[r] = cidx; }
        }
    }

    // ---- reduce (min, idx) across the 16 lanes of each half-wave ----
    // C layout: VGPR r, lanes 0-15 -> row r ; lanes 16-31 -> row r+8.
#pragma unroll
    for (int r = 0; r < 8; ++r) {
        float mv = runmin[r];
        int   mi = runidx[r];
#pragma unroll
        for (int m = 1; m < 16; m <<= 1) {
            float ov = __shfl_xor(mv, m, 32);
            int   oi = __shfl_xor(mi, m, 32);
            if (ov < mv || (ov == mv && oi < mi)) { mv = ov; mi = oi; }
        }
        runmin[r] = mv;
        runidx[r] = mi;
    }

    float loss = zsq;                              // per-lane ||z||^2 piece
    if ((lane & 15) == 0) {                        // lanes 0 and 16
        int rbase = wid * 16 + ((lane >> 4) << 3); // rows 0..7 / 8..15
#pragma unroll
        for (int r = 0; r < 8; ++r) {
            loss += runmin[r];                     // min dist per row (once)
            closestLds[rbase + r] = runidx[r];
        }
    }
#pragma unroll
    for (int m = 1; m < 32; m <<= 1) loss += __shfl_xor(loss, m, 32);
    if (lane == 0) waveLoss[wid] = loss;
    __syncthreads();
    if (threadIdx.x == 0) {
        float s = 0.f;
        for (int w = 0; w < WAVES_PER_BLOCK; ++w) s += waveLoss[w];
        partials[blockIdx.x] = s;
    }

    // ---- gather + write output [B, D, H, W] ----
    // rows n0..n0+15 share (b,h); w = w0 + wofs -> 16 consecutive floats per d.
    const int n0 = tile * 16;
    const int b  = n0 >> 10;                       // / (H*W)
    const int hh = (n0 >> 5) & 31;
    const int w0 = n0 & 31;                        // 0 or 16
    const int wofs = lane & 15;
    const int c = closestLds[wid * 16 + wofs];
    const size_t ob = (size_t)b * (D_ * H_ * W_) + (size_t)hh * W_ + w0 + wofs;
#pragma unroll 4
    for (int j = 0; j < 128; ++j) {
        int d = j * 2 + (lane >> 4);               // lanes<16: even d, else odd
        out[ob + (size_t)d * (H_ * W_)] = weight[d * K_ + c];
    }
}

// Deterministic final loss reduce (double accumulation, fixed order).
__global__ void vq_final(const float* __restrict__ partials,
                         float* __restrict__ out) {
    if (blockIdx.x == 0 && threadIdx.x == 0) {
        double s = 0.0;
        for (int i = 0; i < MAIN_BLOCKS; ++i) s += (double)partials[i];
        float mean = (float)(s / (double)((size_t)N_ * D_));
        out[(size_t)N_ * D_ + 0] = mean;   // commitment_loss
        out[(size_t)N_ * D_ + 1] = mean;   // embedding_loss (same fwd value)
    }
}

extern "C" void kernel_launch(void* const* d_in, const int* in_sizes, int n_in,
                              void* d_out, int out_size, void* d_ws, size_t ws_size,
                              hipStream_t stream) {
    (void)in_sizes; (void)n_in; (void)out_size; (void)ws_size;
    const float* z      = (const float*)d_in[0];
    const float* weight = (const float*)d_in[1];
    float* out = (float*)d_out;

    _Float16* wswz  = (_Float16*)d_ws;
    float* ek2      = (float*)((char*)d_ws + WS_EK2_OFF);
    float* partials = (float*)((char*)d_ws + WS_PART_OFF);

    vq_prep_wswz<<<dim3(1024), dim3(256), 0, stream>>>(weight, wswz);
    vq_prep_ek2 <<<dim3(4),    dim3(256), 0, stream>>>(weight, ek2);
    vq_main     <<<dim3(MAIN_BLOCKS), dim3(256), 0, stream>>>(
        z, weight, wswz, ek2, out, partials);
    vq_final    <<<dim3(1), dim3(32), 0, stream>>>(partials, out);
}
